// CrossAttention_14104672600565
// MI455X (gfx1250) — matrix operand order repaired and checked by reading, not compiled, not run on hardware
//
#include <hip/hip_runtime.h>

typedef __attribute__((ext_vector_type(16))) _Float16 v16h;
typedef __attribute__((ext_vector_type(8)))  float    v8f;
typedef int v4i __attribute__((vector_size(16)));

#define HEADS 8
#define DIMH  64
#define HID   512
#define NBATCH 4
#define SEQ_N1 4096
#define SEQ_N2 1024
#define CH1 256
#define CH2 512
#define ATTN_SCALE 0.125f

#if defined(__has_builtin)
#if __has_builtin(__builtin_amdgcn_global_load_async_to_lds_b128)
#define HAVE_ASYNC_LDS 1
#endif
#endif
#ifndef HAVE_ASYNC_LDS
#define HAVE_ASYNC_LDS 0
#endif

union FragU { uint4 q[2]; v16h h; unsigned u[8]; };

// 16-byte global -> LDS copy. Async (ASYNCcnt-tracked, VGPR-bypassing) when the
// gfx1250 builtin is available; plain load+ds_store staging otherwise.
__device__ __forceinline__ void cp16(const _Float16* g, _Float16* l) {
#if HAVE_ASYNC_LDS
  __builtin_amdgcn_global_load_async_to_lds_b128(
      (__attribute__((address_space(1))) v4i*)(g),
      (__attribute__((address_space(3))) v4i*)(l), 0, 0);
#else
  *(uint4*)l = *(const uint4*)g;
#endif
}

__device__ __forceinline__ void async_wait0() {
#if HAVE_ASYNC_LDS
#if __has_builtin(__builtin_amdgcn_s_wait_asynccnt)
  __builtin_amdgcn_s_wait_asynccnt(0);
#else
  asm volatile("s_wait_asynccnt 0x0" ::: "memory");
#endif
#endif
}

// A-fragment (16x32 f16, M x K): lane L (=lane&15) is row M; lane group g picks
// k-chunk {kb+16g .. kb+16g+15}. Matrix row-major over k.
__device__ __forceinline__ v16h load_fragA(const _Float16* base, int row,
                                           int stride, int kb, int g) {
  const _Float16* p = base + (size_t)row * stride + kb;
  FragU f;
  f.q[0] = *(const uint4*)(p + 16 * g);
  f.q[1] = *(const uint4*)(p + 16 * g + 8);
  return f.h;
}

// B-fragment (32x16 f16, K x N): lane n (=lane&15) is column N; lane group g
// picks k-chunk {kb+16g .. kb+16g+15}. Matrix stored [n][k] (token-major).
__device__ __forceinline__ v16h load_fragB(const _Float16* base, int row,
                                           int stride, int kb, int g) {
  const _Float16* p = base + (size_t)row * stride + kb + 16 * g;
  FragU f;
  f.q[0] = *(const uint4*)(p);
  f.q[1] = *(const uint4*)(p + 8);
  return f.h;
}

__device__ __forceinline__ v8f wmma16(v16h a, v16h b, v8f c) {
  return __builtin_amdgcn_wmma_f32_16x16x32_f16(false, a, false, b, (short)0, c,
                                                false, false);
}

__device__ __forceinline__ unsigned pack2h(float a, float b) {
  union { _Float16 h[2]; unsigned u; } x;
  x.h[0] = (_Float16)a; x.h[1] = (_Float16)b;
  return x.u;
}

__global__ void cvt_f32_f16(const float* __restrict__ s, _Float16* __restrict__ d, int n) {
  int i = blockIdx.x * blockDim.x + threadIdx.x;
  if (i < n) d[i] = (_Float16)s[i];
}

// X [B,C,N] f32 -> XT [B,N,C] f16 via LDS tile (coalesced both sides)
__global__ void transpose_cvt(const float* __restrict__ X, _Float16* __restrict__ XT,
                              int C, int N) {
  __shared__ float tile[16][17];
  int b = blockIdx.z;
  int n0 = blockIdx.x * 16, c0 = blockIdx.y * 16;
  int tx = threadIdx.x, ty = threadIdx.y;
  tile[ty][tx] = X[((size_t)b * C + c0 + ty) * N + n0 + tx];
  __syncthreads();
  XT[((size_t)b * N + n0 + ty) * C + c0 + tx] = (_Float16)tile[tx][ty];
}

// C[m,t] = sum_k A[m,k] * BT[t,k].  Workgroup: 64(M) x 64(N); the 64-token
// B-panel (shared by all 4 waves) is staged once in LDS via async copies.
// MODE 0: store as QT/KT [b,h,n,d] f16 (scaled)    (m -> h*64+d, t -> b,n)
// MODE 1: store as V     [b,h,d,n] f16
// MODE 2: store f32 [b,M,n] + bias[m] + resid      (final projection)
template <int MODE>
__global__ __launch_bounds__(128)
void gemm_tn(const _Float16* __restrict__ A, const _Float16* __restrict__ BT,
             int M, int K, int Ntok, void* __restrict__ dstv, float scale,
             const float* __restrict__ bias, const float* __restrict__ resid) {
  __shared__ _Float16 Bs[64 * 512];  // 64 tokens x K (K <= 512)
  const int tid = threadIdx.x;
  const int lane = tid & 31;
  const int wave = tid >> 5;
  const int L = lane & 15, g = lane >> 4;
  const int m0 = blockIdx.y * 64 + wave * 16;
  const int t0 = blockIdx.x * 64;

  // stage the contiguous 64-row x K panel of BT
  const _Float16* bp = BT + (size_t)t0 * K;
  const int nchunk = (64 * K) >> 3;  // 16B chunks
  for (int c = tid; c < nchunk; c += 128) cp16(bp + c * 8, &Bs[c * 8]);
  async_wait0();
  __syncthreads();

  v8f acc[4] = {};
  for (int k = 0; k < K; k += 32) {
    v16h a = load_fragA(A, m0 + L, K, k, g);
#pragma unroll
    for (int t = 0; t < 4; ++t) {
      v16h b = load_fragB(Bs, t * 16 + L, K, k, g);
      acc[t] = wmma16(a, b, acc[t]);
    }
  }

  const int mrow0 = m0 + 8 * g;  // C-tile: VGPR r -> row mrow0+r, col = lane&15
#pragma unroll
  for (int t = 0; t < 4; ++t) {
    int token = t0 + t * 16 + L;
    int bb = token / Ntok, n = token % Ntok;
    if (MODE == 0) {
      int h = mrow0 >> 6, d0 = mrow0 & 63;
      _Float16* p = (_Float16*)dstv +
                    (((size_t)bb * HEADS + h) * Ntok + n) * DIMH + d0;
      uint4 val;
      val.x = pack2h(acc[t][0] * scale, acc[t][1] * scale);
      val.y = pack2h(acc[t][2] * scale, acc[t][3] * scale);
      val.z = pack2h(acc[t][4] * scale, acc[t][5] * scale);
      val.w = pack2h(acc[t][6] * scale, acc[t][7] * scale);
      *(uint4*)p = val;
    } else if (MODE == 1) {
      _Float16* v = (_Float16*)dstv;
#pragma unroll
      for (int r = 0; r < 8; ++r) {
        int m = mrow0 + r;
        int h = m >> 6, d = m & 63;
        v[(((size_t)bb * HEADS + h) * DIMH + d) * Ntok + n] = (_Float16)acc[t][r];
      }
    } else {
      float* out = (float*)dstv;
#pragma unroll
      for (int r = 0; r < 8; ++r) {
        int c = mrow0 + r;
        size_t idx = ((size_t)bb * M + c) * Ntok + n;
        out[idx] = acc[t][r] + bias[c] + resid[idx];
      }
    }
  }
}

// Flash attention. Workgroup: 64 queries (16/wave); keys consumed in 64-key
// tiles staged in double-buffered LDS via async copies (issue t+1, compute t).
// S^T = (K^T tile) * Q puts queries in lanes, keys across VGPRs -> softmax is
// per-lane + one shfl_xor(16) combine, and the probability C-tile chains into
// the P*V WMMA B-operand with only 4 shfl_xor(16) half-swaps.
__global__ __launch_bounds__(128)
void flash_attn(const _Float16* __restrict__ QT, const _Float16* __restrict__ KT,
                const _Float16* __restrict__ V, _Float16* __restrict__ OT,
                int Nq, int Nk) {
  __shared__ _Float16 Ks[2][64 * DIMH];  // 64 keys x 64 d   (row-major over d)
  __shared__ _Float16 Vs[2][DIMH * 64];  // 64 d   x 64 keys (row-major over keys)
  const int tid = threadIdx.x;
  const int lane = tid & 31;
  const int wave = tid >> 5;
  const int L = lane & 15, g = lane >> 4;
  const int bh = blockIdx.y;
  const int b = bh / HEADS, h = bh % HEADS;
  const int i0 = blockIdx.x * 64 + wave * 16;

  const _Float16* Qb = QT + (size_t)bh * Nq * DIMH;
  const _Float16* Kb = KT + (size_t)bh * Nk * DIMH;
  const _Float16* Vb = V + (size_t)bh * DIMH * Nk;

  auto stage = [&](int jb, int s) {
    // K: 64 rows x 64 halves, contiguous 8KB
#pragma unroll
    for (int i = 0; i < 4; ++i) {
      int c = tid + i * 128;
      cp16(Kb + (size_t)jb * DIMH + c * 8, &Ks[s][c * 8]);
    }
    // V: 64 rows (stride Nk) x 64 halves
#pragma unroll
    for (int i = 0; i < 4; ++i) {
      int c = tid + i * 128;
      int row = c >> 3, cc = c & 7;
      cp16(Vb + (size_t)row * Nk + jb + cc * 8, &Vs[s][row * 64 + cc * 8]);
    }
  };

  const v16h qf0 = load_fragB(Qb, i0 + L, DIMH, 0, g);
  const v16h qf1 = load_fragB(Qb, i0 + L, DIMH, 32, g);

  v8f oacc[4] = {};
  const v8f vzero = {};
  float m_run = -__builtin_inff();
  float l_run = 0.0f;

  stage(0, 0);
  const int ntiles = Nk / 64;
  for (int t = 0; t < ntiles; ++t) {
    async_wait0();
    __syncthreads();  // all waves' copies for tile t visible; prev buffer free
    if (t + 1 < ntiles) stage((t + 1) * 64, (t + 1) & 1);
    const _Float16* Kt = Ks[t & 1];
    const _Float16* Vt = Vs[t & 1];

#pragma unroll
    for (int half = 0; half < 2; ++half) {
      const int jb = half * 32;
      // S^T tiles: Sa = keys jb..jb+15, Sb = keys jb+16..jb+31 (rows), 16 queries
      v16h a0 = load_fragA(Kt, jb + L, DIMH, 0, g);
      v16h a1 = load_fragA(Kt, jb + L, DIMH, 32, g);
      v8f Sa = wmma16(a0, qf0, vzero);
      Sa = wmma16(a1, qf1, Sa);
      a0 = load_fragA(Kt, jb + 16 + L, DIMH, 0, g);
      a1 = load_fragA(Kt, jb + 16 + L, DIMH, 32, g);
      v8f Sb = wmma16(a0, qf0, vzero);
      Sb = wmma16(a1, qf1, Sb);

      // online softmax (per query; lanes l and l^16 hold the same query)
      float mx = -__builtin_inff();
#pragma unroll
      for (int r = 0; r < 8; ++r) { mx = fmaxf(mx, Sa[r]); mx = fmaxf(mx, Sb[r]); }
      mx = fmaxf(mx, __shfl_xor(mx, 16, 32));
      float m_new = fmaxf(m_run, mx);
      float corr = __expf(m_run - m_new);

      float pa[8], pb[8], ls = 0.0f;
#pragma unroll
      for (int r = 0; r < 8; ++r) {
        pa[r] = __expf(Sa[r] - m_new);
        pb[r] = __expf(Sb[r] - m_new);
        ls += pa[r] + pb[r];
      }
      ls += __shfl_xor(ls, 16, 32);
      l_run = l_run * corr + ls;
      m_run = m_new;
#pragma unroll
      for (int dt = 0; dt < 4; ++dt)
#pragma unroll
        for (int r = 0; r < 8; ++r) oacc[dt][r] *= corr;

      // P as B-fragment (k = key, n = query): lane group g needs keys
      // 16g..16g+15; C-layout lane holds keys {8g+r, 16+8g+r} -> swap the
      // mismatched 8-key halves between lane pairs (l, l^16).
      unsigned ha[4], hb[4];
#pragma unroll
      for (int r = 0; r < 4; ++r) {
        ha[r] = pack2h(pa[2 * r], pa[2 * r + 1]);
        hb[r] = pack2h(pb[2 * r], pb[2 * r + 1]);
      }
      FragU pf;
#pragma unroll
      for (int r = 0; r < 4; ++r) {
        unsigned x = __shfl_xor(g ? ha[r] : hb[r], 16, 32);
        pf.u[r]     = g ? x : ha[r];
        pf.u[4 + r] = g ? hb[r] : x;
      }

      // O[d, i] += V[d, jb..jb+31] * P: 4 d-tiles of 16
#pragma unroll
      for (int dt = 0; dt < 4; ++dt) {
        v16h vf = load_fragA(Vt, dt * 16 + L, 64, jb, g);
        oacc[dt] = wmma16(vf, pf.h, oacc[dt]);
      }
    }
  }

  const float inv = 1.0f / l_run;
  const int n = i0 + L;
  _Float16* orow = OT + ((size_t)b * Nq + n) * HID + h * DIMH + 8 * g;
#pragma unroll
  for (int dt = 0; dt < 4; ++dt) {
    uint4 val;
    val.x = pack2h(oacc[dt][0] * inv, oacc[dt][1] * inv);
    val.y = pack2h(oacc[dt][2] * inv, oacc[dt][3] * inv);
    val.z = pack2h(oacc[dt][4] * inv, oacc[dt][5] * inv);
    val.w = pack2h(oacc[dt][6] * inv, oacc[dt][7] * inv);
    *(uint4*)(orow + dt * 16) = val;
  }
}

extern "C" void kernel_launch(void* const* d_in, const int* in_sizes, int n_in,
                              void* d_out, int out_size, void* d_ws, size_t ws_size,
                              hipStream_t stream) {
  const float* seq1 = (const float*)d_in[0];
  const float* seq2 = (const float*)d_in[1];
  const float* Wq1 = (const float*)d_in[2];
  const float* Wk1 = (const float*)d_in[3];
  const float* Wv1 = (const float*)d_in[4];
  const float* Wo1 = (const float*)d_in[5];
  const float* bo1 = (const float*)d_in[6];
  const float* Wq2 = (const float*)d_in[7];
  const float* Wk2 = (const float*)d_in[8];
  const float* Wv2 = (const float*)d_in[9];
  const float* Wo2 = (const float*)d_in[10];
  const float* bo2 = (const float*)d_in[11];

  _Float16* base = (_Float16*)d_ws;
  size_t off = 0;
  auto take = [&](size_t nelem) { _Float16* p = base + off; off += nelem; return p; };
  _Float16* Wq1h = take((size_t)HID * CH1);
  _Float16* Wk1h = take((size_t)HID * CH1);
  _Float16* Wv1h = take((size_t)HID * CH1);
  _Float16* Wo1h = take((size_t)CH1 * HID);
  _Float16* Wq2h = take((size_t)HID * CH2);
  _Float16* Wk2h = take((size_t)HID * CH2);
  _Float16* Wv2h = take((size_t)HID * CH2);
  _Float16* Wo2h = take((size_t)CH2 * HID);
  _Float16* XT1 = take((size_t)NBATCH * SEQ_N1 * CH1);
  _Float16* XT2 = take((size_t)NBATCH * SEQ_N2 * CH2);
  _Float16* QT1 = take((size_t)NBATCH * HID * SEQ_N1);
  _Float16* KT1 = take((size_t)NBATCH * HID * SEQ_N1);
  _Float16* V1  = take((size_t)NBATCH * HID * SEQ_N1);
  _Float16* QT2 = take((size_t)NBATCH * HID * SEQ_N2);
  _Float16* KT2 = take((size_t)NBATCH * HID * SEQ_N2);
  _Float16* V2  = take((size_t)NBATCH * HID * SEQ_N2);
  _Float16* OT1 = take((size_t)NBATCH * SEQ_N1 * HID);
  _Float16* OT2 = take((size_t)NBATCH * SEQ_N2 * HID);

  struct CvJob { const float* s; _Float16* d; int n; };
  CvJob cv[8] = {
      {Wq1, Wq1h, HID * CH1}, {Wk1, Wk1h, HID * CH1}, {Wv1, Wv1h, HID * CH1},
      {Wo1, Wo1h, CH1 * HID}, {Wq2, Wq2h, HID * CH2}, {Wk2, Wk2h, HID * CH2},
      {Wv2, Wv2h, HID * CH2}, {Wo2, Wo2h, CH2 * HID}};
  for (int i = 0; i < 8; ++i)
    cvt_f32_f16<<<(cv[i].n + 255) / 256, 256, 0, stream>>>(cv[i].s, cv[i].d, cv[i].n);

  transpose_cvt<<<dim3(SEQ_N1 / 16, CH1 / 16, NBATCH), dim3(16, 16), 0, stream>>>(
      seq1, XT1, CH1, SEQ_N1);
  transpose_cvt<<<dim3(SEQ_N2 / 16, CH2 / 16, NBATCH), dim3(16, 16), 0, stream>>>(
      seq2, XT2, CH2, SEQ_N2);

  const int T1 = NBATCH * SEQ_N1, T2 = NBATCH * SEQ_N2;
  dim3 blk(128);
  // QKV projections (scale folded into Q)
  gemm_tn<0><<<dim3(T1 / 64, HID / 64), blk, 0, stream>>>(Wq1h, XT1, HID, CH1, SEQ_N1, QT1, ATTN_SCALE, nullptr, nullptr);
  gemm_tn<0><<<dim3(T1 / 64, HID / 64), blk, 0, stream>>>(Wk1h, XT1, HID, CH1, SEQ_N1, KT1, 1.0f, nullptr, nullptr);
  gemm_tn<1><<<dim3(T1 / 64, HID / 64), blk, 0, stream>>>(Wv1h, XT1, HID, CH1, SEQ_N1, V1, 1.0f, nullptr, nullptr);
  gemm_tn<0><<<dim3(T2 / 64, HID / 64), blk, 0, stream>>>(Wq2h, XT2, HID, CH2, SEQ_N2, QT2, ATTN_SCALE, nullptr, nullptr);
  gemm_tn<0><<<dim3(T2 / 64, HID / 64), blk, 0, stream>>>(Wk2h, XT2, HID, CH2, SEQ_N2, KT2, 1.0f, nullptr, nullptr);
  gemm_tn<1><<<dim3(T2 / 64, HID / 64), blk, 0, stream>>>(Wv2h, XT2, HID, CH2, SEQ_N2, V2, 1.0f, nullptr, nullptr);

  // cross attention: q1 attends k2/v2, q2 attends k1/v1
  flash_attn<<<dim3(SEQ_N1 / 64, NBATCH * HEADS), blk, 0, stream>>>(QT1, KT2, V2, OT1, SEQ_N1, SEQ_N2);
  flash_attn<<<dim3(SEQ_N2 / 64, NBATCH * HEADS), blk, 0, stream>>>(QT2, KT1, V1, OT2, SEQ_N2, SEQ_N1);

  // output projections + bias + residual -> f32 d_out (o1 then o2)
  float* o1 = (float*)d_out;
  float* o2 = o1 + (size_t)NBATCH * CH1 * SEQ_N1;
  gemm_tn<2><<<dim3(T1 / 64, CH1 / 64), blk, 0, stream>>>(Wo1h, OT1, CH1, HID, SEQ_N1, o1, 1.0f, bo1, seq1);
  gemm_tn<2><<<dim3(T2 / 64, CH2 / 64), blk, 0, stream>>>(Wo2h, OT2, CH2, HID, SEQ_N2, o2, 1.0f, bo2, seq2);
}
